// GRUCell4RNMT_23768349016090
// MI455X (gfx1250) — matrix operand order repaired and checked by reading, not compiled, not run on hardware
//
#include <hip/hip_runtime.h>
#include <hip/hip_bf16.h>

// ---------------------------------------------------------------------------
// Types for CDNA5 WMMA (wave32): v_wmma_f32_16x16x32_bf16
// ---------------------------------------------------------------------------
typedef __attribute__((ext_vector_type(16))) __bf16 v16bf;
typedef __attribute__((ext_vector_type(8)))  __bf16 v8bf;
typedef __attribute__((ext_vector_type(8)))  float  v8f;

// Explicit global (address_space(1)) pointer types so GEMM loads lower to
// global_load_b128 instead of flat_load_b128.
typedef __attribute__((address_space(1))) const unsigned short gus_t;
typedef __attribute__((address_space(1))) const v8bf           gv8bf_t;
typedef __attribute__((address_space(1))) float                gf32_t;

static __device__ __forceinline__ v8f wmma_bf16(v16bf a, v16bf b, v8f c) {
  // 8 args: (neg_a, A, neg_b, B, c_mod, C, reuse_a, reuse_b)
  return __builtin_amdgcn_wmma_f32_16x16x32_bf16(
      false, a, false, b, (short)0, c, false, false);
}

static __device__ __forceinline__ unsigned short f32_to_bf16(float f) {
  unsigned int u = __float_as_uint(f);
  unsigned int rnd = 0x7FFFu + ((u >> 16) & 1u);   // round-to-nearest-even
  u += rnd;
  return (unsigned short)(u >> 16);
}

// Problem constants (reference is fixed-shape)
#define BATCH 16384
#define IDIM  1024
#define ODIM  1024
#define KDIM  2048          // I + O
#define NDIM  4096          // 2*O (Wt) + O (Wi) + O (Wh)

// LDS staging for B panel: 128 rows x 32 k (bf16), row padded 64B->80B so
// 16-lane ds_load_b128 fragment reads are bank-conflict-free
// (row stride = 20 dwords; banks 20*r mod 64 are 16 disjoint 4-bank groups).
#define B_ROW_PAD   40            // elements per row (32 data + 8 pad)
#define B_BUF_BYTES (128 * B_ROW_PAD * 2)   // 10240 B per buffer

// ---------------------------------------------------------------------------
// Kernel 1: pack A = [x | h] -> bf16 [BATCH x KDIM]
// ---------------------------------------------------------------------------
__global__ __launch_bounds__(256) void pack_a_kernel(
    const float* __restrict__ x, const float* __restrict__ h,
    unsigned short* __restrict__ Abf)
{
  size_t idx = (size_t)blockIdx.x * 256 + threadIdx.x;   // BATCH*KDIM total
  int b = (int)(idx >> 11);
  int c = (int)(idx & 2047);
  float v = (c < IDIM) ? x[(size_t)b * IDIM + c]
                       : h[(size_t)b * ODIM + (c - IDIM)];
  Abf[idx] = f32_to_bf16(v);
}

// ---------------------------------------------------------------------------
// Kernel 2: pack fused weight panel W [NDIM x KDIM] bf16 (row-major, [n][k])
// ---------------------------------------------------------------------------
__global__ __launch_bounds__(256) void pack_w_kernel(
    const float* __restrict__ Wt, const float* __restrict__ Wi,
    const float* __restrict__ Wh, unsigned short* __restrict__ Wbf)
{
  size_t idx = (size_t)blockIdx.x * 256 + threadIdx.x;   // NDIM*KDIM total
  int n = (int)(idx >> 11);
  int k = (int)(idx & 2047);
  float v = 0.0f;
  if (n < 2048) {
    v = Wt[(size_t)n * KDIM + k];
  } else if (n < 3072) {
    if (k < 1024) v = Wi[(size_t)(n - 2048) * IDIM + k];
  } else {
    if (k >= 1024) v = Wh[(size_t)(n - 3072) * ODIM + (k - 1024)];
  }
  Wbf[idx] = f32_to_bf16(v);
}

// ---------------------------------------------------------------------------
// Kernel 3: GEMM  G[M,N] = A[M,K] * W[N,K]^T   (bf16 in, f32 out)
// Block: 256 threads = 8 waves, 4(M) x 2(N); block tile 256x128.
// Wave tile: 64x64 = 4x4 WMMA fragments (128 acc VGPRs).
//
// B panel (128 x 32 per K-step) is staged into LDS with double-buffered
// GLOBAL_LOAD_ASYNC_TO_LDS_B128 (ASYNCcnt), fragments read via ds_load_b128.
// A streams directly from global (L0-resident 2-way reuse) with WGP prefetch.
// ---------------------------------------------------------------------------
__global__ __launch_bounds__(256) void gemm_bf16_kernel(
    const unsigned short* __restrict__ Abf,   // [M,K]
    const unsigned short* __restrict__ Wbf,   // [N,K]
    float* __restrict__ G)                    // [M,N]
{
  __shared__ __align__(16) unsigned short Bsm[2][128 * B_ROW_PAD];

  const int tid  = threadIdx.x;
  const int lane = tid & 31;
  const int wave = tid >> 5;
  const int wm   = wave & 3;                  // wave position along M
  const int wn   = wave >> 2;                 // wave position along N
  const int bm   = blockIdx.x * 256 + wm * 64;
  const int bn0  = blockIdx.y * 128;          // block N origin (region select)

  const int g = lane >> 4;                    // lane K-group
  const int r = lane & 15;                    // M row (A) / N col (B)

  // K-range trimming: padded halves of the fused weight are never touched.
  int kBegin = 0, kEnd = KDIM;
  if (bn0 >= 3072)       kBegin = 1024;       // Wh region: only k >= 1024
  else if (bn0 >= 2048)  kEnd   = 1024;       // Wi region: only k <  1024
  const int kLen = kEnd - kBegin;

  gus_t*  Ag = (gus_t*)(unsigned long long)Abf;
  gf32_t* Gg = (gf32_t*)(unsigned long long)G;

  // --- async B staging setup: thread t moves 2 x 16B chunks per K-step ---
  // chunk c in [0,512): row = c>>2 (0..127), 16B-slot = c&3
  const int brow  = tid >> 1;                 // 0..127
  const int boff0 = (tid & 1) * 2;            // 16B slot 0 or 2
  const int boff1 = boff0 + 1;                // 16B slot 1 or 3
  const unsigned ldsBase =
      (unsigned)(unsigned long long)
      (__attribute__((address_space(3))) unsigned short*)&Bsm[0][0];
  const unsigned lds0 = ldsBase + (unsigned)(brow * B_ROW_PAD + boff0 * 8) * 2u;
  const unsigned lds1 = ldsBase + (unsigned)(brow * B_ROW_PAD + boff1 * 8) * 2u;
  const unsigned long long wBytes = (unsigned long long)Wbf;
  unsigned long long gB0 = wBytes +
      2ull * (unsigned long long)((bn0 + brow) * KDIM + kBegin + boff0 * 8);
  unsigned long long gB1 = wBytes +
      2ull * (unsigned long long)((bn0 + brow) * KDIM + kBegin + boff1 * 8);

  // --- accumulators ---
  v8f acc[4][4];
  #pragma unroll
  for (int i = 0; i < 4; ++i)
    #pragma unroll
    for (int j = 0; j < 4; ++j) { v8f z = {}; acc[i][j] = z; }

  // --- A stream offsets (32-bit element offsets) ---
  int aOff[4];
  #pragma unroll
  for (int i = 0; i < 4; ++i)
    aOff[i] = (bm + 16 * i + r) * KDIM + kBegin + g * 8;

  // Prologue: kick off async fill of buffer 0.
  asm volatile("global_load_async_to_lds_b128 %0, %1, off"
               :: "v"(lds0), "v"(gB0) : "memory");
  asm volatile("global_load_async_to_lds_b128 %0, %1, off"
               :: "v"(lds1), "v"(gB1) : "memory");

  int cur = 0;
  for (int kk = 0; kk < kLen; kk += 32) {
    const bool hasNext = (kk + 32) < kLen;
    if (hasNext) {
      // Issue next K-step into the other buffer (ASYNCcnt += 2).
      const unsigned bufOfs = (unsigned)((cur ^ 1) * B_BUF_BYTES);
      unsigned long long ng0 = gB0 + 2ull * (unsigned long long)(kk + 32);
      unsigned long long ng1 = gB1 + 2ull * (unsigned long long)(kk + 32);
      unsigned nl0 = lds0 + bufOfs, nl1 = lds1 + bufOfs;
      asm volatile("global_load_async_to_lds_b128 %0, %1, off"
                   :: "v"(nl0), "v"(ng0) : "memory");
      asm volatile("global_load_async_to_lds_b128 %0, %1, off"
                   :: "v"(nl1), "v"(ng1) : "memory");
      asm volatile("s_wait_asynccnt 0x2" ::: "memory");  // current buf done
    } else {
      asm volatile("s_wait_asynccnt 0x0" ::: "memory");
    }
    __syncthreads();   // make other waves' async writes visible

    // Prefetch A 4 K-tiles ahead (WGP scope -> pull into near cache).
    __builtin_prefetch((const void*)(Ag + aOff[0] + kk + 128), 0, 3);

    // A fragments: direct global b128 pairs.
    v16bf afrag[4];
    #pragma unroll
    for (int i = 0; i < 4; ++i) {
      v8bf lo = *(gv8bf_t*)(Ag + aOff[i] + kk);        // K: g*8 .. +8
      v8bf hi = *(gv8bf_t*)(Ag + aOff[i] + kk + 16);   // K: 16+g*8 .. +8
      afrag[i] = __builtin_shufflevector(lo, hi,
                   0,1,2,3,4,5,6,7,8,9,10,11,12,13,14,15);
    }
    // B fragments: ds_load_b128 pairs from staged panel.
    v16bf bfrag[4];
    #pragma unroll
    for (int j = 0; j < 4; ++j) {
      const unsigned short* bp =
          &Bsm[cur][(wn * 64 + 16 * j + r) * B_ROW_PAD + g * 8];
      v8bf lo = *(const v8bf*)bp;          // K: g*8 .. +8
      v8bf hi = *(const v8bf*)(bp + 16);   // K: 16+g*8 .. +8
      bfrag[j] = __builtin_shufflevector(lo, hi,
                   0,1,2,3,4,5,6,7,8,9,10,11,12,13,14,15);
    }

    #pragma unroll
    for (int i = 0; i < 4; ++i)
      #pragma unroll
      for (int j = 0; j < 4; ++j)
        acc[i][j] = wmma_bf16(afrag[i], bfrag[j], acc[i][j]);

    __syncthreads();   // all waves done reading buf before it is refilled
    cur ^= 1;
  }

  // Store: lane%16 = N col; VGPR v -> M row = v + 8*(lane/16)
  const int mOff = g * 8;
  const int bn   = bn0 + wn * 64;
  #pragma unroll
  for (int i = 0; i < 4; ++i) {
    #pragma unroll
    for (int j = 0; j < 4; ++j) {
      int cIdx = (bm + 16 * i + mOff) * NDIM + (bn + 16 * j + r);
      #pragma unroll
      for (int v = 0; v < 8; ++v)
        Gg[cIdx + v * NDIM] = acc[i][j][v];
    }
  }
}

// ---------------------------------------------------------------------------
// Kernel 4: per-row LayerNorm over 2048 gate pre-activations + GRU epilogue
// ---------------------------------------------------------------------------
__global__ __launch_bounds__(256) void epilogue_kernel(
    const float* __restrict__ G,      // [BATCH, NDIM]
    const float* __restrict__ state,  // [BATCH, ODIM]
    const float* __restrict__ bi, const float* __restrict__ bh,
    const float* __restrict__ ln_w, const float* __restrict__ ln_b, // [2,ODIM]
    float* __restrict__ out)
{
  const int b = blockIdx.x;
  const float* row = G + (size_t)b * NDIM;

  float s = 0.0f, ss = 0.0f;
  for (int c = threadIdx.x; c < 2 * ODIM; c += 256) {
    float v = row[c];
    s += v; ss += v * v;
  }
  #pragma unroll
  for (int off = 16; off > 0; off >>= 1) {
    s  += __shfl_xor(s, off, 32);
    ss += __shfl_xor(ss, off, 32);
  }
  __shared__ float red_s[8], red_ss[8];
  if ((threadIdx.x & 31) == 0) {
    red_s[threadIdx.x >> 5]  = s;
    red_ss[threadIdx.x >> 5] = ss;
  }
  __syncthreads();
  float tot = 0.0f, tot2 = 0.0f;
  #pragma unroll
  for (int w = 0; w < 8; ++w) { tot += red_s[w]; tot2 += red_ss[w]; }

  const float inv_n = 1.0f / (float)(2 * ODIM);
  float mu   = tot * inv_n;
  float var  = tot2 * inv_n - mu * mu;
  float rstd = rsqrtf(var + 1e-6f);

  for (int o = threadIdx.x; o < ODIM; o += 256) {
    float n0 = (row[o]        - mu) * rstd * ln_w[o]        + ln_b[o];
    float n1 = (row[ODIM + o] - mu) * rstd * ln_w[ODIM + o] + ln_b[ODIM + o];
    float ig = 1.0f / (1.0f + __expf(-n0));
    float fg = 1.0f / (1.0f + __expf(-n1));
    float xw = row[2 * ODIM + o];       // x @ Wi.T
    float hw = row[3 * ODIM + o];       // h @ Wh.T
    float hidden = tanhf(xw + bi[o] + ig * (hw + bh[o]));
    float st = state[(size_t)b * ODIM + o];
    out[(size_t)b * ODIM + o] = (1.0f - fg) * hidden + fg * st;
  }
}

// ---------------------------------------------------------------------------
// Host launcher
// ---------------------------------------------------------------------------
extern "C" void kernel_launch(void* const* d_in, const int* in_sizes, int n_in,
                              void* d_out, int out_size, void* d_ws, size_t ws_size,
                              hipStream_t stream) {
  const float* inpute = (const float*)d_in[0];   // [B, I]
  const float* state  = (const float*)d_in[1];   // [B, O]
  const float* Wt     = (const float*)d_in[2];   // [2O, I+O]
  const float* Wi     = (const float*)d_in[3];   // [O, I]
  const float* bi     = (const float*)d_in[4];   // [O]
  const float* Wh     = (const float*)d_in[5];   // [O, O]
  const float* bh     = (const float*)d_in[6];   // [O]
  const float* ln_w   = (const float*)d_in[7];   // [2, O]
  const float* ln_b   = (const float*)d_in[8];   // [2, O]
  float* out = (float*)d_out;

  // Workspace layout
  char* ws = (char*)d_ws;
  unsigned short* Abf = (unsigned short*)ws;                       // 64 MB
  unsigned short* Wbf = (unsigned short*)(ws + (size_t)BATCH * KDIM * 2);
  float* G = (float*)(ws + (size_t)BATCH * KDIM * 2
                         + (size_t)NDIM * KDIM * 2);               // 256 MB

  {
    size_t total = (size_t)BATCH * KDIM;
    pack_a_kernel<<<dim3((unsigned)(total / 256)), dim3(256), 0, stream>>>(
        inpute, state, Abf);
  }
  {
    size_t total = (size_t)NDIM * KDIM;
    pack_w_kernel<<<dim3((unsigned)(total / 256)), dim3(256), 0, stream>>>(
        Wt, Wi, Wh, Wbf);
  }
  {
    dim3 grid(BATCH / 256, NDIM / 128);
    gemm_bf16_kernel<<<grid, dim3(256), 0, stream>>>(Abf, Wbf, G);
  }
  {
    epilogue_kernel<<<dim3(BATCH), dim3(256), 0, stream>>>(
        G, state, bi, bh, ln_w, ln_b, out);
  }
}
